// EnhancedAttentionModule_1168231104866
// MI455X (gfx1250) — matrix operand order repaired and checked by reading, not compiled, not run on hardware
//
#include <hip/hip_runtime.h>

typedef __attribute__((ext_vector_type(16))) _Float16 v16h;
typedef __attribute__((ext_vector_type(8)))  float    v8f;
typedef __attribute__((ext_vector_type(4)))  _Float16 v4h;

#define BB 16
#define CC 512
#define CR 128
#define NN 4096
#define NT 64     // n-columns per block in fused kernel
#define LS 520    // padded LDS row stride in halves: 1040B rows -> 16B aligned, dword stride 260 (conflict-free)

// ---------------------------------------------------------------------------
// Kernel 1: pooled[b,c] = mean_n x[b,c,n].  One wave (32 lanes) per row.
// ---------------------------------------------------------------------------
__global__ __launch_bounds__(256)
void pool_kernel(const float* __restrict__ x, float* __restrict__ pooled)
{
    const int tid  = threadIdx.x;
    const int lane = tid & 31;
    const int wave = tid >> 5;
    const int row  = blockIdx.x * 8 + wave;          // 0 .. B*C-1
    const float* base = x + (size_t)row * NN;

    float4 s4 = make_float4(0.f, 0.f, 0.f, 0.f);
    #pragma unroll 4
    for (int i = 0; i < NN / 128; ++i) {             // 32 iters, 128 floats/wave/iter
        float4 v = *reinterpret_cast<const float4*>(base + i * 128 + lane * 4);
        s4.x += v.x; s4.y += v.y; s4.z += v.z; s4.w += v.w;
    }
    float s = (s4.x + s4.y) + (s4.z + s4.w);
    #pragma unroll
    for (int off = 16; off >= 1; off >>= 1)
        s += __shfl_xor(s, off, 32);
    if (lane == 0) pooled[row] = s * (1.f / (float)NN);
}

// ---------------------------------------------------------------------------
// Kernel 2: one-time w3 f32 -> f16 conversion (feeds the WMMA A-matrix).
// ---------------------------------------------------------------------------
__global__ __launch_bounds__(256)
void w3cvt_kernel(const float* __restrict__ w3, _Float16* __restrict__ w3h)
{
    int i = blockIdx.x * 256 + threadIdx.x;          // 0 .. CR*CC/4-1
    float4 v = reinterpret_cast<const float4*>(w3)[i];
    v4h h;
    h.x = (_Float16)v.x; h.y = (_Float16)v.y;
    h.z = (_Float16)v.z; h.w = (_Float16)v.w;
    reinterpret_cast<v4h*>(w3h)[i] = h;
}

// ---------------------------------------------------------------------------
// Kernel 3 (1 block): channel-attention MLP -> ca[B,C]; fold b3+BN -> scale/bias.
// ---------------------------------------------------------------------------
__global__ __launch_bounds__(256)
void mlp_kernel(const float* __restrict__ pooled,
                const float* __restrict__ w1, const float* __restrict__ b1,
                const float* __restrict__ w2, const float* __restrict__ b2,
                const float* __restrict__ b3,
                const float* __restrict__ g,  const float* __restrict__ be,
                const float* __restrict__ mu, const float* __restrict__ var,
                float* __restrict__ ca, float* __restrict__ bns, float* __restrict__ bnb)
{
    __shared__ float ps[BB * CC];    // 32 KB
    __shared__ float hs[BB * CR];    // 8 KB
    const int tid = threadIdx.x;

    for (int i = tid; i < BB * CC; i += 256) ps[i] = pooled[i];
    __syncthreads();

    // h = relu(pooled @ w1^T + b1)
    for (int i = tid; i < BB * CR; i += 256) {
        int bi = i >> 7, o = i & (CR - 1);
        float s = b1[o];
        const float* wr = w1 + o * CC;
        const float* pr = ps + bi * CC;
        #pragma unroll 8
        for (int c = 0; c < CC; ++c) s = fmaf(wr[c], pr[c], s);
        hs[i] = fmaxf(s, 0.f);
    }
    __syncthreads();

    // ca = sigmoid(h @ w2^T + b2)
    for (int i = tid; i < BB * CC; i += 256) {
        int bi = i >> 9, c = i & (CC - 1);
        float s = b2[c];
        const float* wr = w2 + c * CR;
        const float* hr = hs + bi * CR;
        #pragma unroll 8
        for (int k = 0; k < CR; ++k) s = fmaf(wr[k], hr[k], s);
        ca[i] = 1.f / (1.f + __expf(-s));
    }

    // fold conv bias b3 and BN running stats into scale/bias
    if (tid < CR) {
        float inv = g[tid] * rsqrtf(var[tid] + 1e-5f);
        bns[tid] = inv;
        bnb[tid] = (b3[tid] - mu[tid]) * inv + be[tid];
    }
}

// ---------------------------------------------------------------------------
// Kernel 4: fused spatial attention + output for one (batch, 64-col) tile.
//   h2 = w3 @ (x*ca) via v_wmma_f32_16x16x32_f16 (A streamed from L2,
//   prefetched one K-step ahead; B from LDS), then BN+relu+w4 reduction,
//   sa = sigmoid(...), out = x * (1 + ca*sa).
//   LDS ~70 KB -> 4 blocks/WGP -> 8 waves/SIMD to hide LDS/L2 latency.
// ---------------------------------------------------------------------------
__global__ __launch_bounds__(256)
void fused_kernel(const float* __restrict__ x, const _Float16* __restrict__ w3h,
                  const float* __restrict__ cag, const float* __restrict__ bnsg,
                  const float* __restrict__ bnbg, const float* __restrict__ w4,
                  const float* __restrict__ b4, float* __restrict__ out)
{
    extern __shared__ char smem[];
    _Float16* xs = reinterpret_cast<_Float16*>(smem);     // [NT][LS]  (x*ca, transposed, f16)
    float* cas  = reinterpret_cast<float*>(xs + NT * LS); // [CC]
    float* w4s  = cas + CC;                               // [CR]
    float* bns  = w4s + CR;                               // [CR]
    float* bnb  = bns + CR;                               // [CR]
    float* sacc = bnb + CR;                               // [NT]

    const int tid  = threadIdx.x;
    const int lane = tid & 31;
    const int wave = tid >> 5;
    const int b    = blockIdx.y;
    const int n0   = blockIdx.x * NT;

    // ---- stage small per-block arrays ----
    for (int i = tid; i < CC; i += 256) cas[i] = cag[b * CC + i];
    for (int i = tid; i < CR; i += 256) { w4s[i] = w4[i]; bns[i] = bnsg[i]; bnb[i] = bnbg[i]; }
    if (tid < NT) sacc[tid] = 0.f;
    __syncthreads();

    // ---- x tile * ca -> xs[n][c] f16 (transpose so B-fragments are contiguous in K) ----
    for (int idx = tid; idx < CC * (NT / 4); idx += 256) {
        int c  = idx >> 4;
        int j4 = (idx & 15) << 2;
        float4 v = *reinterpret_cast<const float4*>(x + ((size_t)b * CC + c) * NN + n0 + j4);
        float cc = cas[c];
        xs[(j4 + 0) * LS + c] = (_Float16)(v.x * cc);
        xs[(j4 + 1) * LS + c] = (_Float16)(v.y * cc);
        xs[(j4 + 2) * LS + c] = (_Float16)(v.z * cc);
        xs[(j4 + 3) * LS + c] = (_Float16)(v.w * cc);
    }
    __syncthreads();

    // ---- GEMM: each wave -> M-tile = wave (16 rows of Cr), 4 N-subtiles ----
    // A layout (ISA 16-bit A 16x32): lane<16 -> K 0-7 & 16-23, lane>=16 -> K 8-15 & 24-31
    // B layout (matches A): lane<16 -> K 0-7 & 16-23, lane>=16 -> K 8-15 & 24-31
    const int mrow = wave * 16 + (lane & 15);
    const int aklo = (lane & 16) ? 8 : 0;
    const int bklo = (lane & 16) ? 8 : 0;
    const int bn   = lane & 15;

    const _Float16* arow = w3h + (size_t)mrow * CC;   // global (L2-resident, 16 KB/wave/block)

    v8f acc[4] = {};
    v16h af_cur, af_nxt;
    // prologue: A fragment for k = 0
    *reinterpret_cast<float4*>(&af_cur)       = *reinterpret_cast<const float4*>(arow + aklo);
    *(reinterpret_cast<float4*>(&af_cur) + 1) = *reinterpret_cast<const float4*>(arow + 16 + aklo);

    for (int k = 0; k < CC; k += 32) {
        // prefetch A for next K-step (clamped re-read on the last step)
        const int kn = (k + 32 < CC) ? (k + 32) : k;
        *reinterpret_cast<float4*>(&af_nxt)       = *reinterpret_cast<const float4*>(arow + kn + aklo);
        *(reinterpret_cast<float4*>(&af_nxt) + 1) = *reinterpret_cast<const float4*>(arow + kn + 16 + aklo);

        #pragma unroll
        for (int nt = 0; nt < 4; ++nt) {
            const _Float16* brow = xs + (nt * 16 + bn) * LS + k + bklo;
            v16h bf;
            *reinterpret_cast<float4*>(&bf)       = *reinterpret_cast<const float4*>(brow);
            *(reinterpret_cast<float4*>(&bf) + 1) = *reinterpret_cast<const float4*>(brow + 16);
            acc[nt] = __builtin_amdgcn_wmma_f32_16x16x32_f16(
                false, af_cur, false, bf, (short)0, acc[nt], false, false);
        }
        af_cur = af_nxt;
    }

    // ---- epilogue: BN + relu + w4 partial reduction ----
    // accumulator VGPR r: M = r (lane<16) / r+8 (lane>=16), N = lane&15
    const int obase = wave * 16 + ((lane & 16) ? 8 : 0);
    #pragma unroll
    for (int nt = 0; nt < 4; ++nt) {
        float s = 0.f;
        #pragma unroll
        for (int r = 0; r < 8; ++r) {
            int o = obase + r;
            float h = fmaf(acc[nt][r], bns[o], bnb[o]);
            h = fmaxf(h, 0.f);
            s = fmaf(h, w4s[o], s);
        }
        s += __shfl_xor(s, 16, 32);                  // merge the two M-halves (same N)
        if (lane < 16) atomicAdd(&sacc[nt * 16 + bn], s);   // ds_add_f32 across waves
    }
    __syncthreads();

    if (tid < NT) {
        float v = sacc[tid] + b4[0];
        sacc[tid] = 1.f / (1.f + __expf(-v));        // sa[n]
    }
    __syncthreads();

    // ---- out = x * (1 + ca*sa), float4 stores (x re-read hits WGP$/L2) ----
    for (int idx = tid; idx < CC * (NT / 4); idx += 256) {
        int c  = idx >> 4;
        int j4 = (idx & 15) << 2;
        size_t base = ((size_t)b * CC + c) * NN + n0 + j4;
        float4 v = *reinterpret_cast<const float4*>(x + base);
        float cc = cas[c];
        float4 o;
        o.x = v.x * fmaf(cc, sacc[j4 + 0], 1.f);
        o.y = v.y * fmaf(cc, sacc[j4 + 1], 1.f);
        o.z = v.z * fmaf(cc, sacc[j4 + 2], 1.f);
        o.w = v.w * fmaf(cc, sacc[j4 + 3], 1.f);
        *reinterpret_cast<float4*>(out + base) = o;
    }
}

// ---------------------------------------------------------------------------
extern "C" void kernel_launch(void* const* d_in, const int* in_sizes, int n_in,
                              void* d_out, int out_size, void* d_ws, size_t ws_size,
                              hipStream_t stream)
{
    const float* x   = (const float*)d_in[0];
    const float* w1  = (const float*)d_in[1];
    const float* b1  = (const float*)d_in[2];
    const float* w2  = (const float*)d_in[3];
    const float* b2  = (const float*)d_in[4];
    const float* w3  = (const float*)d_in[5];
    const float* b3  = (const float*)d_in[6];
    const float* g   = (const float*)d_in[7];
    const float* be  = (const float*)d_in[8];
    const float* mu  = (const float*)d_in[9];
    const float* var = (const float*)d_in[10];
    const float* w4  = (const float*)d_in[11];
    const float* b4  = (const float*)d_in[12];
    float* out = (float*)d_out;

    // workspace carve
    float* pooled = (float*)d_ws;            // B*C
    float* ca     = pooled + BB * CC;        // B*C
    float* bns    = ca + BB * CC;            // CR
    float* bnb    = bns + CR;                // CR
    _Float16* w3h = (_Float16*)(bnb + CR);   // CR*CC halves (16B-aligned offset)

    pool_kernel <<<dim3(BB * CC / 8), dim3(256), 0, stream>>>(x, pooled);
    w3cvt_kernel<<<dim3(CR * CC / (256 * 4)), dim3(256), 0, stream>>>(w3, w3h);
    mlp_kernel  <<<dim3(1), dim3(256), 0, stream>>>(pooled, w1, b1, w2, b2, b3,
                                                    g, be, mu, var, ca, bns, bnb);

    const size_t smem = (size_t)(NT * LS) * sizeof(_Float16)
                      + (size_t)(CC + 3 * CR + NT) * sizeof(float);   // ~70 KB -> 4 blocks/WGP
    fused_kernel<<<dim3(NN / NT, BB), dim3(256), smem, stream>>>(
        x, w3h, ca, bns, bnb, w4, b4, out);
}
